// VAE_62672162783507
// MI455X (gfx1250) — matrix operand order repaired and checked by reading, not compiled, not run on hardware
//
#include <hip/hip_runtime.h>

// ============================================================================
// PointNet++-style VAE forward for MI455X (gfx1250, wave32, WMMA).
//
// Grouped shared-MLPs + dense chain run on v_wmma_f32_16x16x32_f16 with f32
// accumulation. SA weights are pre-converted once to zero-padded f16 (they are
// reused ~16K times); dense weights stay fp32 in HBM (one-shot, bandwidth
// bound) and are converted in-register while building fragments.
//
// Assumed d_in flattening (jax pytree order):
//   0 obj_pc (32,7,1024)   1 eps (32,64)
//   2..7   dec[0..2]  {W,b}
//   8..13  enc[0..2]  {W,b}
//   14,15  fc2 {W,b}        16,17 fc2_bn {b,s}
//   18,19  lat_fc0 {W,b}    20,21 lat_fc1 {W,b}
//   22,23  lat_logvar {W,b} 24,25 lat_mean {W,b}
//   26     lat_sc
//   27..35 sa1[0..2] {W,b,s}   36..44 sa2   45..53 sa3
// ============================================================================

typedef __attribute__((ext_vector_type(16))) _Float16 v16h;
typedef __attribute__((ext_vector_type(8)))  float    v8f;

constexpr int cmax2(int a, int b) { return a > b ? a : b; }

__device__ __forceinline__ v8f wmma_f16(v16h a, v16h b, v8f c) {
  // D = A(16x32 f16) * B(32x16 f16) + C(16x16 f32)
  return __builtin_amdgcn_wmma_f32_16x16x32_f16(false, a, false, b,
                                                (short)0, c, false, false);
}

// A fragment from an f16 row-major tile (LDS). row0 = this wave's M-tile row 0.
// Element e: k = kBase + (l/16)*16 + e.  Two 16B ds loads.
__device__ __forceinline__ v16h load_a_lds(const _Float16* row0, int ld,
                                           int kBase, int lane) {
  const int m = lane & 15, h = lane >> 4;
  const _Float16* q = row0 + (size_t)m * ld + kBase + h * 16;
  v16h a;
#pragma unroll
  for (int e = 0; e < 16; ++e) a[e] = q[e];
  return a;
}

// A fragment from an fp32 row-major matrix in global memory (dense chain).
// 16 unconditional f32 loads -> 4x global_load_b128 + v_cvt_pk_f16_f32.
__device__ __forceinline__ v16h load_a_gf32(const float* row0, int ld,
                                            int kBase, int lane) {
  const int m = lane & 15, h = lane >> 4;
  const float* q = row0 + (size_t)m * ld + kBase + h * 16;
  v16h a;
#pragma unroll
  for (int e = 0; e < 16; ++e) a[e] = (_Float16)q[e];
  return a;
}

// B fragment from pre-converted, zero-padded f16 weights (ld = padded K).
// Lane n = n0 + l%16, element e <-> k = kBase + (l/16)*16 + e.
// 16 contiguous halves -> two 16B vector loads, no branches.
__device__ __forceinline__ v16h load_b_h(const _Float16* Wh, int ld, int n0,
                                         int kBase, int lane) {
  const _Float16* p = Wh + (size_t)(n0 + (lane & 15)) * ld + kBase + (lane >> 4) * 16;
  v16h b;
#pragma unroll
  for (int e = 0; e < 16; ++e) b[e] = p[e];
  return b;
}

// B fragment from fp32 weights, NO bounds check (K multiple of 32, full rows).
__device__ __forceinline__ v16h load_b_f32(const float* W, int ld, int n0,
                                           int kBase, int lane) {
  const float* p = W + (size_t)(n0 + (lane & 15)) * ld + kBase + (lane >> 4) * 16;
  v16h b;
#pragma unroll
  for (int e = 0; e < 16; ++e) b[e] = (_Float16)p[e];
  return b;
}

// ---------------------------------------------------------------------------
// Weight convert: W (Cout x K fp32) -> Wh (Cout x KP f16), zero pad k >= K.
// ---------------------------------------------------------------------------
__global__ __launch_bounds__(256) void k_cvt_w(const float* __restrict__ W,
                                               _Float16* __restrict__ Wh,
                                               int Cout, int K, int KP) {
  const int t = blockIdx.x * 256 + threadIdx.x;
  if (t >= Cout * KP) return;
  const int n = t / KP, k = t - n * KP;
  Wh[t] = (k < K) ? (_Float16)W[(size_t)n * K + k] : (_Float16)0.f;
}

// ---------------------------------------------------------------------------
// split obj_pc (B,7,N) -> xyz (B,N,3), feat (B,N,4)
// ---------------------------------------------------------------------------
__global__ __launch_bounds__(256) void k_split(const float* __restrict__ pc,
                                               float* __restrict__ xyz,
                                               float* __restrict__ feat) {
  const int t = blockIdx.x * 256 + threadIdx.x;
  if (t >= 32 * 1024) return;
  const int b = t >> 10, n = t & 1023;
  const float* src = pc + (size_t)b * 7 * 1024 + n;
#pragma unroll
  for (int k = 0; k < 3; ++k) xyz[(size_t)t * 3 + k] = src[(size_t)k * 1024];
#pragma unroll
  for (int c = 0; c < 4; ++c) feat[(size_t)t * 4 + c] = src[(size_t)(3 + c) * 1024];
}

// ---------------------------------------------------------------------------
// Farthest point sampling, one block per batch, LDS-resident points+dists.
// ---------------------------------------------------------------------------
__global__ __launch_bounds__(256) void k_fps(const float* __restrict__ xyz,
                                             int N, int npoint,
                                             int* __restrict__ fidx,
                                             float* __restrict__ newxyz) {
  __shared__ float sx[1024], sy[1024], sz[1024], sdist[1024];
  __shared__ float rv[256];
  __shared__ int   ri[256];
  __shared__ int   scur;
  const int b = blockIdx.x, tid = threadIdx.x;
  for (int n = tid; n < N; n += 256) {
    const float* p = xyz + ((size_t)b * N + n) * 3;
    sx[n] = p[0]; sy[n] = p[1]; sz[n] = p[2];
    sdist[n] = 1e10f;
  }
  if (tid == 0) scur = 0;
  __syncthreads();
  for (int it = 0; it < npoint; ++it) {
    const int cur = scur;
    if (tid == 0) {
      fidx[(size_t)b * npoint + it] = cur;
      float* q = newxyz + ((size_t)b * npoint + it) * 3;
      q[0] = sx[cur]; q[1] = sy[cur]; q[2] = sz[cur];
    }
    const float cx = sx[cur], cy = sy[cur], cz = sz[cur];
    float best = -1.f; int bidx = 0;
    for (int n = tid; n < N; n += 256) {
      const float dx = sx[n] - cx, dy = sy[n] - cy, dz = sz[n] - cz;
      const float d = fminf(sdist[n], dx * dx + dy * dy + dz * dz);
      sdist[n] = d;
      if (d > best) { best = d; bidx = n; }
    }
    rv[tid] = best; ri[tid] = bidx;
    __syncthreads();
    for (int off = 128; off > 0; off >>= 1) {
      if (tid < off) {
        if (rv[tid + off] > rv[tid] ||
            (rv[tid + off] == rv[tid] && ri[tid + off] < ri[tid])) {
          rv[tid] = rv[tid + off]; ri[tid] = ri[tid + off];
        }
      }
      __syncthreads();
    }
    if (tid == 0) scur = ri[0];
    __syncthreads();
  }
}

// ---------------------------------------------------------------------------
// Ball query: one wave per query point, ascending-index compaction via ballot.
// ---------------------------------------------------------------------------
__global__ __launch_bounds__(256) void k_ballquery(
    const float* __restrict__ xyz, const float* __restrict__ newxyz,
    int N, int S, int BS, float r2, int nsample, int* __restrict__ idx) {
  const int gtid = blockIdx.x * 256 + threadIdx.x;
  const int wid = gtid >> 5;
  const int lane = gtid & 31;
  if (wid >= BS) return;
  const int b = wid / S, s = wid % S;
  const float* q = newxyz + ((size_t)b * S + s) * 3;
  const float qx = q[0], qy = q[1], qz = q[2];
  int* op = idx + ((size_t)b * S + s) * nsample;
  int cnt = 0, firstJ = -1;
  for (int j0 = 0; j0 < N && cnt < nsample; j0 += 32) {
    const int j = j0 + lane;
    bool pred = false;
    if (j < N) {
      const float* p = xyz + ((size_t)b * N + j) * 3;
      const float dx = p[0] - qx, dy = p[1] - qy, dz = p[2] - qz;
      pred = (dx * dx + dy * dy + dz * dz) < r2;
    }
    const unsigned mask = (unsigned)__ballot(pred ? 1 : 0);
    if (pred) {
      const int pos = cnt + __popc(mask & ((1u << lane) - 1u));
      if (pos < nsample) op[pos] = j;
    }
    if (firstJ < 0 && mask) firstJ = j0 + __ffs(mask) - 1;
    cnt += __popc(mask);
  }
  if (cnt > nsample) cnt = nsample;
  if (firstJ < 0) firstJ = 0;
  for (int t = cnt + lane; t < nsample; t += 32) op[t] = firstJ;
}

// Epilogue of one SA 16x16 tile: relu(acc*s[n]+b[n]) -> f16 LDS tile.
__device__ __forceinline__ void sa_epi(v8f acc, _Float16* dst, int ld, int m0,
                                       int n, const float* sc, const float* bi) {
  const float s = sc[n], b = bi[n];
#pragma unroll
  for (int i = 0; i < 8; ++i) {
    float t = acc[i] * s + b;
    t = fmaxf(t, 0.f);
    dst[(size_t)(m0 + i) * ld + n] = (_Float16)t;
  }
}

// ---------------------------------------------------------------------------
// One shared-MLP layer, dual N-tile WMMA chains sharing one A fragment.
// src/dst row stride CMAX halves; Wh (COUT x KP f16, padded).
// ---------------------------------------------------------------------------
template <int KP, int COUT, int CMAX>
__device__ __forceinline__ void mlp_layer(const _Float16* __restrict__ src,
                                          _Float16* __restrict__ dst,
                                          const _Float16* __restrict__ Wh,
                                          const float* __restrict__ sc,
                                          const float* __restrict__ bi,
                                          int wave, int lane) {
  const _Float16* srow = src + (size_t)(wave * 16) * CMAX;
  const int m0 = wave * 16 + (lane >> 4) * 8;
  const int nl = lane & 15;
#pragma unroll 1
  for (int nt = 0; nt < COUT / 16; nt += 2) {
    v8f acc0 = {}, acc1 = {};
#pragma unroll 1
    for (int kc = 0; kc < KP; kc += 32) {
      const v16h a  = load_a_lds(srow, CMAX, kc, lane);
      const v16h b0 = load_b_h(Wh, KP, nt * 16, kc, lane);
      const v16h b1 = load_b_h(Wh, KP, nt * 16 + 16, kc, lane);
      acc0 = wmma_f16(a, b0, acc0);
      acc1 = wmma_f16(a, b1, acc1);
    }
    sa_epi(acc0, dst, CMAX, m0, nt * 16 + nl, sc, bi);
    sa_epi(acc1, dst, CMAX, m0, nt * 16 + 16 + nl, sc, bi);
  }
}

// ---------------------------------------------------------------------------
// Set-abstraction: gather/group -> 3x (conv1x1 + bn + relu) via WMMA -> max.
// idxp==nullptr => group_all mode (rows = points s*NS+si, no recentering).
// ---------------------------------------------------------------------------
template <int CIN, int C1, int C2, int C3, int NS>
__global__ __launch_bounds__(NS * 2) void k_sa_mlp(
    const float* __restrict__ xyz, const float* __restrict__ feat,
    const float* __restrict__ newxyz, const int* __restrict__ idxp,
    int N, int S,
    const _Float16* __restrict__ W0, const float* __restrict__ s0, const float* __restrict__ b0,
    const _Float16* __restrict__ W1, const float* __restrict__ s1, const float* __restrict__ b1,
    const _Float16* __restrict__ W2, const float* __restrict__ s2, const float* __restrict__ b2,
    float* __restrict__ out) {
  constexpr int KP0 = (CIN + 31) & ~31;
  constexpr int CMAX = cmax2(cmax2(KP0, C1), cmax2(C2, C3));
  constexpr int BD = NS * 2;
  __shared__ _Float16 bufA[NS * CMAX];
  __shared__ _Float16 bufB[NS * CMAX];
  const int tid = threadIdx.x;
  const int lane = tid & 31;
  const int wave = tid >> 5;
  const int b = blockIdx.x / S;
  const int s = blockIdx.x % S;

  // Build grouped f16 input tile (NS x KP0 at stride CMAX), zero-padded K.
  for (int t = tid; t < NS * KP0; t += BD) {
    const int si = t / KP0, k = t - si * KP0;
    int j;
    if (idxp) j = idxp[((size_t)b * S + s) * NS + si];
    else      j = s * NS + si;
    float v = 0.f;
    if (k < 3) {
      v = xyz[((size_t)b * N + j) * 3 + k];
      if (newxyz) v -= newxyz[((size_t)b * S + s) * 3 + k];
    } else if (k < CIN) {
      v = feat[((size_t)b * N + j) * (CIN - 3) + (k - 3)];
    }
    bufA[(size_t)si * CMAX + k] = (_Float16)v;
  }
  __syncthreads();
  mlp_layer<KP0, C1, CMAX>(bufA, bufB, W0, s0, b0, wave, lane);
  __syncthreads();
  mlp_layer<C1, C2, CMAX>(bufB, bufA, W1, s1, b1, wave, lane);
  __syncthreads();
  mlp_layer<C2, C3, CMAX>(bufA, bufB, W2, s2, b2, wave, lane);
  __syncthreads();
  // Max over the NS samples of the group.
  for (int c = tid; c < C3; c += BD) {
    float mx = (float)bufB[c];
    for (int m = 1; m < NS; ++m) mx = fmaxf(mx, (float)bufB[(size_t)m * CMAX + c]);
    out[((size_t)b * S + s) * C3 + c] = mx;
  }
}

// Dense epilogue for one 16x16 tile: all configuration compile-time.
template <int ACT, bool HB, bool HS>
__device__ __forceinline__ void dense_epi(v8f acc, int n, int m0,
                                          const float* __restrict__ bias1,
                                          const float* __restrict__ scale,
                                          const float* __restrict__ bias2,
                                          float* __restrict__ Y, int Nout) {
  const float b1v = HB ? bias1[n] : 0.f;
  const float sv = HS ? scale[n] : 1.f;
  const float b2v = HS ? bias2[n] : 0.f;
#pragma unroll
  for (int i = 0; i < 8; ++i) {
    float t = (acc[i] + b1v) * sv + b2v;
    if (ACT == 1) t = fmaxf(t, 0.f);
    else if (ACT == 2) t = 1.f / (1.f + __expf(-t));
    Y[(size_t)(m0 + i) * Nout + n] = t;
  }
}

// ---------------------------------------------------------------------------
// Dense: Y(32,Nout) = act((X*W^T + b1)*s + b2). One wave per 16x32 tile pair.
// grid = (Nout/32, 2). ACT: 0 none, 1 relu, 2 sigmoid. K % 32 == 0.
// ---------------------------------------------------------------------------
template <int ACT, bool HB, bool HS>
__global__ __launch_bounds__(32) void k_dense(
    const float* __restrict__ X, const float* __restrict__ W,
    const float* __restrict__ bias1, const float* __restrict__ scale,
    const float* __restrict__ bias2, float* __restrict__ Y,
    int K, int Nout) {
  const int lane = threadIdx.x;
  const int n0 = blockIdx.x * 32, mt = blockIdx.y;
  const float* Xr = X + (size_t)mt * 16 * K;
  v8f acc0 = {}, acc1 = {};
  for (int kc = 0; kc < K; kc += 32) {
    const v16h a  = load_a_gf32(Xr, K, kc, lane);
    const v16h b0 = load_b_f32(W, K, n0, kc, lane);
    const v16h b1 = load_b_f32(W, K, n0 + 16, kc, lane);
    acc0 = wmma_f16(a, b0, acc0);
    acc1 = wmma_f16(a, b1, acc1);
  }
  const int nl = lane & 15;
  const int m0 = mt * 16 + (lane >> 4) * 8;
  dense_epi<ACT, HB, HS>(acc0, n0 + nl, m0, bias1, scale, bias2, Y, Nout);
  dense_epi<ACT, HB, HS>(acc1, n0 + 16 + nl, m0, bias1, scale, bias2, Y, Nout);
}

// Partial-group max combine for sa3 (B,4,256) -> (B,256).
__global__ __launch_bounds__(256) void k_gmax(const float* __restrict__ part,
                                              float* __restrict__ out) {
  const int t = blockIdx.x * 256 + threadIdx.x;
  if (t >= 32 * 256) return;
  const int b = t >> 8, c = t & 255;
  float mx = part[((size_t)b * 4 + 0) * 256 + c];
#pragma unroll
  for (int s = 1; s < 4; ++s) mx = fmaxf(mx, part[((size_t)b * 4 + s) * 256 + c]);
  out[t] = mx;
}

// hb = leaky_relu(x_s + dx, 0.2)
__global__ __launch_bounds__(256) void k_hb(const float* __restrict__ xs,
                                            const float* __restrict__ dx,
                                            float* __restrict__ hb, int n) {
  const int i = blockIdx.x * 256 + threadIdx.x;
  if (i >= n) return;
  const float t = xs[i] + dx[i];
  hb[i] = t > 0.f ? t : 0.2f * t;
}

// z = eps * exp(0.5*logvar) + mu
__global__ __launch_bounds__(256) void k_z(const float* __restrict__ mu,
                                           const float* __restrict__ lv,
                                           const float* __restrict__ eps,
                                           float* __restrict__ zo,
                                           float* __restrict__ zw, int n) {
  const int i = blockIdx.x * 256 + threadIdx.x;
  if (i >= n) return;
  const float z = eps[i] * __expf(0.5f * lv[i]) + mu[i];
  zo[i] = z; zw[i] = z;
}

// ============================================================================
extern "C" void kernel_launch(void* const* d_in, const int* in_sizes, int n_in,
                              void* d_out, int out_size, void* d_ws, size_t ws_size,
                              hipStream_t stream) {
  (void)in_sizes; (void)n_in; (void)out_size; (void)ws_size;
  const int B = 32;

  const float* obj_pc = (const float*)d_in[0];
  const float* eps    = (const float*)d_in[1];
  const float* dec0W = (const float*)d_in[2],  *dec0b = (const float*)d_in[3];
  const float* dec1W = (const float*)d_in[4],  *dec1b = (const float*)d_in[5];
  const float* dec2W = (const float*)d_in[6],  *dec2b = (const float*)d_in[7];
  const float* enc0W = (const float*)d_in[8],  *enc0b = (const float*)d_in[9];
  const float* enc1W = (const float*)d_in[10], *enc1b = (const float*)d_in[11];
  const float* enc2W = (const float*)d_in[12], *enc2b = (const float*)d_in[13];
  const float* fc2W  = (const float*)d_in[14], *fc2b  = (const float*)d_in[15];
  const float* bnb   = (const float*)d_in[16], *bns   = (const float*)d_in[17];
  const float* l0W   = (const float*)d_in[18], *l0b   = (const float*)d_in[19];
  const float* l1W   = (const float*)d_in[20], *l1b   = (const float*)d_in[21];
  const float* lvW   = (const float*)d_in[22], *lvb   = (const float*)d_in[23];
  const float* muW   = (const float*)d_in[24], *mub   = (const float*)d_in[25];
  const float* scW   = (const float*)d_in[26];
  const float* sa1W[3] = {(const float*)d_in[27], (const float*)d_in[30], (const float*)d_in[33]};
  const float* sa1b[3] = {(const float*)d_in[28], (const float*)d_in[31], (const float*)d_in[34]};
  const float* sa1s[3] = {(const float*)d_in[29], (const float*)d_in[32], (const float*)d_in[35]};
  const float* sa2W[3] = {(const float*)d_in[36], (const float*)d_in[39], (const float*)d_in[42]};
  const float* sa2b[3] = {(const float*)d_in[37], (const float*)d_in[40], (const float*)d_in[43]};
  const float* sa2s[3] = {(const float*)d_in[38], (const float*)d_in[41], (const float*)d_in[44]};
  const float* sa3W[3] = {(const float*)d_in[45], (const float*)d_in[48], (const float*)d_in[51]};
  const float* sa3b[3] = {(const float*)d_in[46], (const float*)d_in[49], (const float*)d_in[52]};
  const float* sa3s[3] = {(const float*)d_in[47], (const float*)d_in[50], (const float*)d_in[53]};

  // ---- workspace carve-out (256B aligned) ----
  char* wp = (char*)d_ws;
  auto allocF = [&](size_t n) { float* r = (float*)wp; wp += ((n * 4 + 255) & ~(size_t)255); return r; };
  auto allocI = [&](size_t n) { int*   r = (int*)wp;   wp += ((n * 4 + 255) & ~(size_t)255); return r; };
  auto allocH = [&](size_t n) { _Float16* r = (_Float16*)wp; wp += ((n * 2 + 255) & ~(size_t)255); return r; };
  float* xyz  = allocF((size_t)B * 1024 * 3);
  float* feat = allocF((size_t)B * 1024 * 4);
  int*   fid1 = allocI((size_t)B * 512);
  float* nx1  = allocF((size_t)B * 512 * 3);
  int*   idx1 = allocI((size_t)B * 512 * 64);
  float* f1   = allocF((size_t)B * 512 * 128);
  int*   fid2 = allocI((size_t)B * 128);
  float* nx2  = allocF((size_t)B * 128 * 3);
  int*   idx2 = allocI((size_t)B * 128 * 64);
  float* f2   = allocF((size_t)B * 128 * 256);
  float* bpart= allocF((size_t)B * 4 * 256);
  float* bott = allocF((size_t)B * 256);
  float* g    = allocF((size_t)B * 4096);
  float* e0   = allocF((size_t)B * 2048);
  float* e1   = allocF((size_t)B * 1024);
  float* h    = allocF((size_t)B * 256);
  float* net  = allocF((size_t)B * 128);
  float* dxv  = allocF((size_t)B * 128);
  float* xsv  = allocF((size_t)B * 128);
  float* hb   = allocF((size_t)B * 128);
  float* zws  = allocF((size_t)B * 64);
  float* dd0  = allocF((size_t)B * 128);
  float* dd1  = allocF((size_t)B * 256);
  // SA weights, f16 zero-padded to K multiple of 32.
  _Float16* sa1Wh[3] = {allocH(64 * 32), allocH(64 * 64), allocH(128 * 64)};
  _Float16* sa2Wh[3] = {allocH(128 * 160), allocH(128 * 128), allocH(256 * 128)};
  _Float16* sa3Wh[3] = {allocH(256 * 288), allocH(256 * 256), allocH(256 * 256)};

  float* out  = (float*)d_out;          // recon(32768) | mu(2048) | lv(2048) | z(2048)
  float* muO  = out + 32768;
  float* lvO  = out + 34816;
  float* zO   = out + 36864;

  // ---- SA weight conversion (tiny, reused by thousands of blocks) ----
  {
    const float* Wf[9] = {sa1W[0], sa1W[1], sa1W[2], sa2W[0], sa2W[1], sa2W[2],
                          sa3W[0], sa3W[1], sa3W[2]};
    _Float16* Wh[9] = {sa1Wh[0], sa1Wh[1], sa1Wh[2], sa2Wh[0], sa2Wh[1], sa2Wh[2],
                       sa3Wh[0], sa3Wh[1], sa3Wh[2]};
    const int Co[9] = {64, 64, 128, 128, 128, 256, 256, 256, 256};
    const int Kk[9] = {7, 64, 64, 131, 128, 128, 259, 256, 256};
    const int Kp[9] = {32, 64, 64, 160, 128, 128, 288, 256, 256};
    for (int i = 0; i < 9; ++i) {
      const int tot = Co[i] * Kp[i];
      k_cvt_w<<<(tot + 255) / 256, 256, 0, stream>>>(Wf[i], Wh[i], Co[i], Kk[i], Kp[i]);
    }
  }

  // ---- point branch ----
  k_split<<<(B * 1024) / 256, 256, 0, stream>>>(obj_pc, xyz, feat);
  k_fps<<<B, 256, 0, stream>>>(xyz, 1024, 512, fid1, nx1);
  k_ballquery<<<(B * 512 * 32) / 256, 256, 0, stream>>>(xyz, nx1, 1024, 512, B * 512, 0.04f, 64, idx1);
  k_sa_mlp<7, 64, 64, 128, 64><<<B * 512, 128, 0, stream>>>(
      xyz, feat, nx1, idx1, 1024, 512,
      sa1Wh[0], sa1s[0], sa1b[0], sa1Wh[1], sa1s[1], sa1b[1], sa1Wh[2], sa1s[2], sa1b[2], f1);
  k_fps<<<B, 256, 0, stream>>>(nx1, 512, 128, fid2, nx2);
  k_ballquery<<<(B * 128 * 32) / 256, 256, 0, stream>>>(nx1, nx2, 512, 128, B * 128, 0.16f, 64, idx2);
  k_sa_mlp<131, 128, 128, 256, 64><<<B * 128, 128, 0, stream>>>(
      nx1, f1, nx2, idx2, 512, 128,
      sa2Wh[0], sa2s[0], sa2b[0], sa2Wh[1], sa2s[1], sa2b[1], sa2Wh[2], sa2s[2], sa2b[2], f2);
  // group_all: 128 points split into 4 sub-groups of 32 rows, max combined after.
  k_sa_mlp<259, 256, 256, 256, 32><<<B * 4, 64, 0, stream>>>(
      nx2, f2, nullptr, nullptr, 128, 4,
      sa3Wh[0], sa3s[0], sa3b[0], sa3Wh[1], sa3s[1], sa3b[1], sa3Wh[2], sa3s[2], sa3b[2], bpart);
  k_gmax<<<(B * 256) / 256, 256, 0, stream>>>(bpart, bott);

  // ---- dense chain (M = 32) ----
  k_dense<1, true, true><<<dim3(4096 / 32, 2), 32, 0, stream>>>(bott, fc2W, fc2b, bns, bnb, g, 256, 4096);
  k_dense<1, true, false><<<dim3(2048 / 32, 2), 32, 0, stream>>>(g, enc0W, enc0b, nullptr, nullptr, e0, 4096, 2048);
  k_dense<1, true, false><<<dim3(1024 / 32, 2), 32, 0, stream>>>(e0, enc1W, enc1b, nullptr, nullptr, e1, 2048, 1024);
  k_dense<1, true, false><<<dim3(256 / 32, 2), 32, 0, stream>>>(e1, enc2W, enc2b, nullptr, nullptr, h, 1024, 256);
  // net only feeds relu(net): fuse the relu here.
  k_dense<1, true, false><<<dim3(128 / 32, 2), 32, 0, stream>>>(h, l0W, l0b, nullptr, nullptr, net, 256, 128);
  k_dense<0, true, false><<<dim3(128 / 32, 2), 32, 0, stream>>>(net, l1W, l1b, nullptr, nullptr, dxv, 128, 128);
  k_dense<0, false, false><<<dim3(128 / 32, 2), 32, 0, stream>>>(h, scW, nullptr, nullptr, nullptr, xsv, 256, 128);
  k_hb<<<(B * 128 + 255) / 256, 256, 0, stream>>>(xsv, dxv, hb, B * 128);
  k_dense<0, true, false><<<dim3(64 / 32, 2), 32, 0, stream>>>(hb, muW, mub, nullptr, nullptr, muO, 128, 64);
  k_dense<0, true, false><<<dim3(64 / 32, 2), 32, 0, stream>>>(hb, lvW, lvb, nullptr, nullptr, lvO, 128, 64);
  k_z<<<(B * 64 + 255) / 256, 256, 0, stream>>>(muO, lvO, eps, zO, zws, B * 64);
  k_dense<1, true, false><<<dim3(128 / 32, 2), 32, 0, stream>>>(zws, dec0W, dec0b, nullptr, nullptr, dd0, 64, 128);
  k_dense<1, true, false><<<dim3(256 / 32, 2), 32, 0, stream>>>(dd0, dec1W, dec1b, nullptr, nullptr, dd1, 128, 256);
  k_dense<2, true, false><<<dim3(1024 / 32, 2), 32, 0, stream>>>(dd1, dec2W, dec2b, nullptr, nullptr, out, 256, 1024);
}
